// HaloAttn_28982439313895
// MI455X (gfx1250) — matrix operand order repaired and checked by reading, not compiled, not run on hardware
//
#include <hip/hip_runtime.h>

// ---------------------------------------------------------------------------
// Halo attention on MI455X (gfx1250): all GEMM-shaped work through
// v_wmma_f32_16x16x32_bf16 (wave32 WMMA), bf16 intermediates to halve HBM
// traffic, f32 accumulation. Async global->LDS (ASYNCcnt) used where data is
// already bf16 and needs no conversion.
// ---------------------------------------------------------------------------

typedef __bf16 bf16_t;
typedef __attribute__((ext_vector_type(16))) __bf16 v16bf;
typedef __attribute__((ext_vector_type(8)))  float  v8f;
typedef __attribute__((ext_vector_type(4)))  int    v4i;

union BF16x16 { v16bf v; unsigned int u[8]; };
union F32x8   { v8f   v; float f[8]; };
union U32BF2  { unsigned int u; bf16_t h[2]; };

#ifndef __has_builtin
#define __has_builtin(x) 0
#endif
#if __has_builtin(__builtin_amdgcn_global_load_async_to_lds_b128)
#define HAVE_ASYNC_LDS 1
#else
#define HAVE_ASYNC_LDS 0
#endif

// Builtin signature (from hipcc diagnostic): (v4i AS1*, v4i AS3*, Ii, Ii)
typedef __attribute__((address_space(1))) v4i as1_v4i;
typedef __attribute__((address_space(3))) v4i as3_v4i;

// 16-byte global -> LDS copy; async (no VGPR round trip) when available.
__device__ __forceinline__ void copy16_g2lds(void* lds, const void* g) {
#if HAVE_ASYNC_LDS
    __builtin_amdgcn_global_load_async_to_lds_b128(
        (as1_v4i*)g, (as3_v4i*)lds, /*offset=*/0, /*cpol=*/0);
#else
    *(uint4*)lds = *(const uint4*)g;
#endif
}

__device__ __forceinline__ void wait_async_lds() {
#if HAVE_ASYNC_LDS
    asm volatile("s_wait_asynccnt 0x0" ::: "memory");
#endif
}

// Constants for this problem instance
#define IMG_H   128
#define IMG_W   128
#define CH      256
#define NHEADS  8
#define DHEAD   32          // dqk == dv
#define BSX     8           // block size
#define HALO    3
#define WINSZ   14          // BSX + 2*HALO
#define NTOK    196         // WINSZ*WINSZ
#define WPAD    224         // NTOK padded to multiple of 32
#define QTOK    64          // BSX*BSX

// K-pair pattern for 16-bit A fragment (16x32), matching B fills:
// lanes 0-15 hold K {0..15}; lanes 16-31 hold K {16..31},
// packed 2 per VGPR.
__device__ __forceinline__ int kpatA(int j, int lane) {
    return ((lane >> 4) << 4) + (j << 1);
}

__device__ __forceinline__ v8f wmma_bf16(const BF16x16& a, const BF16x16& b, v8f c) {
    return __builtin_amdgcn_wmma_f32_16x16x32_bf16(
        /*neg_a=*/false, a.v, /*neg_b=*/false, b.v,
        /*c_mod=*/(short)0, c, /*reuse_a=*/false, /*reuse_b=*/false);
}

// ---------------------------------------------------------------------------
// Kernel 1: qkv = x @ [Wq | Wkv]   (M=65536, N=768, K=256), output bf16
// grid (1024, 12), block 128 (4 waves). 64x64 tile per WG, 16x64 per wave.
// ---------------------------------------------------------------------------
__global__ void __launch_bounds__(128)
qkv_gemm(const float* __restrict__ x, const float* __restrict__ Wq,
         const float* __restrict__ Wkv, bf16_t* __restrict__ qkv) {
    __shared__ bf16_t As[64 * 32];   // A tile, row-major, stride 32
    __shared__ bf16_t Bt[64 * 32];   // B tile transposed: [n][k], stride 32

    const int m0 = blockIdx.x * 64;
    const int n0 = blockIdx.y * 64;
    const int tid = threadIdx.x;
    const int lane = tid & 31;
    const int wave = tid >> 5;

    F32x8 acc[4];
    for (int t = 0; t < 4; ++t)
        for (int j = 0; j < 8; ++j) acc[t].f[j] = 0.f;

    for (int k0 = 0; k0 < CH; k0 += 32) {
        // --- stage A (f32 -> bf16), 64x32: batch loads, then convert+store
        float2 xr[8];
#pragma unroll
        for (int u = 0; u < 8; ++u) {
            const int i = tid + u * 128;
            const int m = i >> 4;
            const int kp = (i & 15) << 1;
            xr[u] = *(const float2*)&x[(size_t)(m0 + m) * CH + k0 + kp];
        }
#pragma unroll
        for (int u = 0; u < 8; ++u) {
            const int i = tid + u * 128;
            const int m = i >> 4;
            const int kp = (i & 15) << 1;
            As[m * 32 + kp]     = (bf16_t)xr[u].x;
            As[m * 32 + kp + 1] = (bf16_t)xr[u].y;
        }
        // --- stage B transposed (f32 -> bf16), 32x64 -> Bt[n][k]
        float wr[16];
#pragma unroll
        for (int u = 0; u < 16; ++u) {
            const int i = tid + u * 128;
            const int kk = i >> 6;
            const int n  = i & 63;
            const int gn = n0 + n;
            wr[u] = (gn < 256) ? Wq[(size_t)(k0 + kk) * 256 + gn]
                               : Wkv[(size_t)(k0 + kk) * 512 + (gn - 256)];
        }
#pragma unroll
        for (int u = 0; u < 16; ++u) {
            const int i = tid + u * 128;
            const int kk = i >> 6;
            const int n  = i & 63;
            Bt[n * 32 + kk] = (bf16_t)wr[u];
        }
        __syncthreads();

        // prefetch next A chunk into cache
        if (k0 + 32 < CH)
            __builtin_prefetch(&x[(size_t)(m0 + wave * 16) * CH + k0 + 32], 0, 1);

        BF16x16 afr;
        const int am = wave * 16 + (lane & 15);
#pragma unroll
        for (int j = 0; j < 8; ++j)
            afr.u[j] = *(const unsigned int*)&As[am * 32 + kpatA(j, lane)];

        const int kb = ((lane >> 4) & 1) * 16;
#pragma unroll
        for (int nt = 0; nt < 4; ++nt) {
            BF16x16 bfr;
            const int bn = nt * 16 + (lane & 15);
#pragma unroll
            for (int j = 0; j < 8; ++j)
                bfr.u[j] = *(const unsigned int*)&Bt[bn * 32 + kb + 2 * j];
            acc[nt].v = wmma_bf16(afr, bfr, acc[nt].v);
        }
        __syncthreads();
    }

    // store D (bf16)
#pragma unroll
    for (int nt = 0; nt < 4; ++nt) {
        const int col = n0 + nt * 16 + (lane & 15);
#pragma unroll
        for (int vv = 0; vv < 8; ++vv) {
            const int row = m0 + wave * 16 + vv + ((lane >> 4) & 1) * 8;
            qkv[(size_t)row * 768 + col] = (bf16_t)acc[nt].f[vv];
        }
    }
}

// ---------------------------------------------------------------------------
// Kernel 2: halo attention. One WG per (batch, head, block).
// S = (Q*scale) @ K^T  -> masked softmax -> O = P @ V
// ---------------------------------------------------------------------------
__global__ void __launch_bounds__(128)
halo_attn(const bf16_t* __restrict__ qkv, bf16_t* __restrict__ attn_out) {
    __shared__ bf16_t Kw[WPAD * DHEAD];   // [tok][d], stride 32
    __shared__ bf16_t Vt[DHEAD * WPAD];   // [d][tok], stride 224
    __shared__ float  Sf[QTOK * WPAD];    // scores f32
    __shared__ bf16_t Pm[QTOK * WPAD];    // probabilities bf16 (A-frag friendly)

    const int bid  = blockIdx.x;
    const int b    = bid >> 11;          // / (NHEADS*256)
    const int rem  = bid & 2047;
    const int head = rem >> 8;
    const int blk  = rem & 255;
    const int by   = blk >> 4;
    const int bx   = blk & 15;

    const int tid  = threadIdx.x;
    const int lane = tid & 31;
    const int wave = tid >> 5;

    const unsigned int* qkv32 = (const unsigned int*)qkv;

    // ---- stage K window: async b128 for valid tokens, zero-fill invalid.
    // Each token's K row = 32 bf16 = 4 x 16B chunks; disjoint LDS addresses
    // between the zero (DS) and async (ASYNCcnt) streams.
    for (int idx = tid; idx < WPAD * 4; idx += 128) {
        const int t = idx >> 2;           // window token 0..223
        const int c = idx & 3;            // 16B chunk (8 bf16)
        const int wy = t / WINSZ;
        const int wx = t - wy * WINSZ;
        const int py = by * BSX - HALO + wy;
        const int px = bx * BSX - HALO + wx;
        const bool valid =
            (t < NTOK) && ((unsigned)py < IMG_H) && ((unsigned)px < IMG_W);
        bf16_t* dst = &Kw[t * DHEAD + c * 8];
        if (valid) {
            const bf16_t* src =
                qkv + (size_t)((b * IMG_H + py) * IMG_W + px) * 768 + 256 +
                head * 64 + c * 8;
            copy16_g2lds(dst, src);
        } else {
            uint4 z = {0u, 0u, 0u, 0u};
            *(uint4*)dst = z;
        }
    }

    // ---- stage V window transposed into Vt[d][tok] (manual: needs transpose)
    for (int idx = tid; idx < WPAD * 16; idx += 128) {
        const int t  = idx >> 4;          // token
        const int ch = idx & 15;          // u32 chunk of V row (2 bf16)
        const int wy = t / WINSZ;
        const int wx = t - wy * WINSZ;
        const int py = by * BSX - HALO + wy;
        const int px = bx * BSX - HALO + wx;
        const bool valid =
            (t < NTOK) && ((unsigned)py < IMG_H) && ((unsigned)px < IMG_W);
        const int d0 = ch << 1;
        unsigned int val = 0u;
        if (valid) {
            const size_t base =
                (size_t)((b * IMG_H + py) * IMG_W + px) * 768 + 256 +
                head * 64 + DHEAD + d0;
            val = qkv32[base >> 1];
        }
        U32BF2 u; u.u = val;
        Vt[d0 * WPAD + t]       = u.h[0];
        Vt[(d0 + 1) * WPAD + t] = u.h[1];
    }

    // ---- Q fragment straight from global (16x32 per wave)
    BF16x16 aq;
    {
        const int qrow = wave * 16 + (lane & 15);
        const int iy = qrow >> 3, ix = qrow & 7;
        const int y = by * BSX + iy, x = bx * BSX + ix;
        const size_t base =
            (size_t)((b * IMG_H + y) * IMG_W + x) * 768 + head * DHEAD;
#pragma unroll
        for (int j = 0; j < 8; ++j)
            aq.u[j] = qkv32[(base + kpatA(j, lane)) >> 1];
    }
    wait_async_lds();
    __syncthreads();

    // ---- S = Q @ K^T (14 n-tiles of 16 window tokens)
    const float scale = 0.17677669529663687f; // 1/sqrt(32)
    const int kb = ((lane >> 4) & 1) * 16;
#pragma unroll
    for (int nt = 0; nt < WPAD / 16; ++nt) {
        BF16x16 bk;
        const int n = nt * 16 + (lane & 15);
#pragma unroll
        for (int j = 0; j < 8; ++j)
            bk.u[j] = *(const unsigned int*)&Kw[n * DHEAD + kb + 2 * j];
        F32x8 s;
        for (int j = 0; j < 8; ++j) s.f[j] = 0.f;
        s.v = wmma_bf16(aq, bk, s.v);
        const int col = nt * 16 + (lane & 15);
#pragma unroll
        for (int vv = 0; vv < 8; ++vv) {
            const int row = wave * 16 + vv + ((lane >> 4) & 1) * 8;
            Sf[row * WPAD + col] = s.f[vv] * scale;
        }
    }
    __syncthreads();

    // ---- masked softmax: 2 lanes per query row
    {
        const int r = tid >> 1;
        const int p = tid & 1;
        float mx = -3.4e38f;
        for (int c = p; c < WPAD; c += 2) {
            const int wy = c / WINSZ, wx = c - wy * WINSZ;
            const int py = by * BSX - HALO + wy, px = bx * BSX - HALO + wx;
            const bool valid =
                (c < NTOK) && ((unsigned)py < IMG_H) && ((unsigned)px < IMG_W);
            if (valid) mx = fmaxf(mx, Sf[r * WPAD + c]);
        }
        mx = fmaxf(mx, __shfl_xor(mx, 1, 32));
        float sum = 0.f;
        for (int c = p; c < WPAD; c += 2) {
            const int wy = c / WINSZ, wx = c - wy * WINSZ;
            const int py = by * BSX - HALO + wy, px = bx * BSX - HALO + wx;
            const bool valid =
                (c < NTOK) && ((unsigned)py < IMG_H) && ((unsigned)px < IMG_W);
            if (valid) sum += __expf(Sf[r * WPAD + c] - mx);
        }
        sum += __shfl_xor(sum, 1, 32);
        const float inv = 1.0f / sum;
        for (int c = p; c < WPAD; c += 2) {
            const int wy = c / WINSZ, wx = c - wy * WINSZ;
            const int py = by * BSX - HALO + wy, px = bx * BSX - HALO + wx;
            const bool valid =
                (c < NTOK) && ((unsigned)py < IMG_H) && ((unsigned)px < IMG_W);
            const float pv = valid ? __expf(Sf[r * WPAD + c] - mx) * inv : 0.f;
            Pm[r * WPAD + c] = (bf16_t)pv;
        }
    }
    __syncthreads();

    // ---- O = P @ V  (K loop 7 chunks of 32 tokens, 2 n-tiles of head dim)
    F32x8 o[2];
    for (int t = 0; t < 2; ++t)
        for (int j = 0; j < 8; ++j) o[t].f[j] = 0.f;

#pragma unroll
    for (int kc = 0; kc < WPAD; kc += 32) {
        BF16x16 ap;
        const int m = wave * 16 + (lane & 15);
#pragma unroll
        for (int j = 0; j < 8; ++j)
            ap.u[j] = *(const unsigned int*)&Pm[m * WPAD + kc + kpatA(j, lane)];
#pragma unroll
        for (int nt = 0; nt < 2; ++nt) {
            BF16x16 bv;
            const int d = nt * 16 + (lane & 15);
#pragma unroll
            for (int j = 0; j < 8; ++j)
                bv.u[j] = *(const unsigned int*)&Vt[d * WPAD + kc + kb + 2 * j];
            o[nt].v = wmma_bf16(ap, bv, o[nt].v);
        }
    }

    // ---- store attention output (bf16, [B,H,W, head*32+d])
#pragma unroll
    for (int nt = 0; nt < 2; ++nt) {
        const int d = nt * 16 + (lane & 15);
#pragma unroll
        for (int vv = 0; vv < 8; ++vv) {
            const int qr = wave * 16 + vv + ((lane >> 4) & 1) * 8;
            const int iy = qr >> 3, ix = qr & 7;
            const int y = by * BSX + iy, x = bx * BSX + ix;
            attn_out[(size_t)((b * IMG_H + y) * IMG_W + x) * CH +
                     head * DHEAD + d] = (bf16_t)o[nt].f[vv];
        }
    }
}

// ---------------------------------------------------------------------------
// Kernel 3: out = attn(bf16) @ Wproj + bias   (M=65536, N=256, K=256), f32 out
// grid (1024, 4), block 128. A tile staged with async global->LDS.
// ---------------------------------------------------------------------------
__global__ void __launch_bounds__(128)
proj_gemm(const bf16_t* __restrict__ ain, const float* __restrict__ Wp,
          const float* __restrict__ bias, float* __restrict__ out) {
    __shared__ bf16_t As[64 * 32];
    __shared__ bf16_t Bt[64 * 32];

    const int m0 = blockIdx.x * 64;
    const int n0 = blockIdx.y * 64;
    const int tid = threadIdx.x;
    const int lane = tid & 31;
    const int wave = tid >> 5;

    F32x8 acc[4];
    for (int t = 0; t < 4; ++t)
        for (int j = 0; j < 8; ++j) acc[t].f[j] = 0.f;

    for (int k0 = 0; k0 < CH; k0 += 32) {
        // A tile: bf16 already -> async 16B copies (2 per thread)
#pragma unroll
        for (int u = 0; u < 2; ++u) {
            const int i = tid + u * 128;   // 0..255
            const int m = i >> 2;
            const int c = i & 3;
            copy16_g2lds(&As[m * 32 + c * 8],
                         ain + (size_t)(m0 + m) * CH + k0 + c * 8);
        }
        // B tile: f32 -> bf16 transposed, batched loads
        float wr[16];
#pragma unroll
        for (int u = 0; u < 16; ++u) {
            const int i = tid + u * 128;
            const int kk = i >> 6;
            const int n  = i & 63;
            wr[u] = Wp[(size_t)(k0 + kk) * CH + (n0 + n)];
        }
#pragma unroll
        for (int u = 0; u < 16; ++u) {
            const int i = tid + u * 128;
            const int kk = i >> 6;
            const int n  = i & 63;
            Bt[n * 32 + kk] = (bf16_t)wr[u];
        }
        wait_async_lds();
        __syncthreads();

        BF16x16 afr;
        const int am = wave * 16 + (lane & 15);
#pragma unroll
        for (int j = 0; j < 8; ++j)
            afr.u[j] = *(const unsigned int*)&As[am * 32 + kpatA(j, lane)];

        const int kb = ((lane >> 4) & 1) * 16;
#pragma unroll
        for (int nt = 0; nt < 4; ++nt) {
            BF16x16 bfr;
            const int bn = nt * 16 + (lane & 15);
#pragma unroll
            for (int j = 0; j < 8; ++j)
                bfr.u[j] = *(const unsigned int*)&Bt[bn * 32 + kb + 2 * j];
            acc[nt].v = wmma_bf16(afr, bfr, acc[nt].v);
        }
        __syncthreads();
    }

#pragma unroll
    for (int nt = 0; nt < 4; ++nt) {
        const int col = n0 + nt * 16 + (lane & 15);
        const float bb = bias[col];
#pragma unroll
        for (int vv = 0; vv < 8; ++vv) {
            const int row = m0 + wave * 16 + vv + ((lane >> 4) & 1) * 8;
            out[(size_t)row * CH + col] = acc[nt].f[vv] + bb;
        }
    }
}

// ---------------------------------------------------------------------------
extern "C" void kernel_launch(void* const* d_in, const int* in_sizes, int n_in,
                              void* d_out, int out_size, void* d_ws, size_t ws_size,
                              hipStream_t stream) {
    const float* x     = (const float*)d_in[0];
    const float* Wq    = (const float*)d_in[1];
    const float* Wkv   = (const float*)d_in[2];
    const float* Wproj = (const float*)d_in[3];
    const float* bproj = (const float*)d_in[4];

    const size_t npix = (size_t)4 * IMG_H * IMG_W;          // 65536
    bf16_t* qkv  = (bf16_t*)d_ws;                            // [npix][768]
    bf16_t* attn = qkv + npix * 768;                         // [npix][256]
    float*  out  = (float*)d_out;

    dim3 blk(128);
    qkv_gemm<<<dim3(1024, 12), blk, 0, stream>>>(x, Wq, Wkv, qkv);
    halo_attn<<<dim3(4 * NHEADS * 256), blk, 0, stream>>>(qkv, attn);
    proj_gemm<<<dim3(1024, 4), blk, 0, stream>>>(attn, Wproj, bproj, out);
}
